// RwkvSelfAttention_6803228196957
// MI455X (gfx1250) — matrix operand order repaired and checked by reading, not compiled, not run on hardware
//
#include <hip/hip_runtime.h>

#define B_ 4
#define T_ 2048
#define H_ 2048
#define M_ (B_ * T_)   // 8192 GEMM rows
#define S_ 16          // scan segments
#define L_ (T_ / S_)   // 128 tokens per segment

typedef __attribute__((ext_vector_type(16))) __bf16        v16bf;
typedef __attribute__((ext_vector_type(8)))  float         v8f;
typedef __attribute__((ext_vector_type(4)))  unsigned int  v4u;

union Frag16 { v16bf v; v4u u[2]; };

// ---------------------------------------------------------------------------
// Weight transpose + fp32 -> bf16 convert:  Wt[n][k] = bf16(W[k][n])
// ---------------------------------------------------------------------------
__global__ void __launch_bounds__(256)
wt_transpose_kernel(const float* __restrict__ W0, const float* __restrict__ W1,
                    const float* __restrict__ W2, const float* __restrict__ W3,
                    __bf16* __restrict__ T0, __bf16* __restrict__ T1,
                    __bf16* __restrict__ T2, __bf16* __restrict__ T3)
{
    __shared__ float tile[32][33];
    const float* W; __bf16* Wt;
    switch (blockIdx.z) {
        case 0:  W = W0; Wt = T0; break;
        case 1:  W = W1; Wt = T1; break;
        case 2:  W = W2; Wt = T2; break;
        default: W = W3; Wt = T3; break;
    }
    const int n  = blockIdx.x * 32 + threadIdx.x;   // source column
    const int k0 = blockIdx.y * 32;                 // source row base
    for (int j = threadIdx.y; j < 32; j += 8)
        tile[j][threadIdx.x] = W[(size_t)(k0 + j) * H_ + n];
    __syncthreads();
    const int k  = k0 + threadIdx.x;
    const int n0 = blockIdx.x * 32;
    for (int j = threadIdx.y; j < 32; j += 8)
        Wt[(size_t)(n0 + j) * H_ + k] = (__bf16)tile[threadIdx.x][j];
}

// ---------------------------------------------------------------------------
// Token-shift mix:  x*tm + shifted*(1-tm), emitted as bf16 GEMM A-matrices
// ---------------------------------------------------------------------------
__global__ void __launch_bounds__(256)
mix_kernel(const float* __restrict__ hidden, const float* __restrict__ sx,
           const float* __restrict__ tmk, const float* __restrict__ tmv,
           const float* __restrict__ tmr,
           __bf16* __restrict__ mixK, __bf16* __restrict__ mixV,
           __bf16* __restrict__ mixR)
{
    const size_t i = (size_t)blockIdx.x * blockDim.x + threadIdx.x;  // over B*T*H
    const int h  = (int)(i % H_);
    const size_t bt = i / H_;
    const int t  = (int)(bt % T_);
    const int b  = (int)(bt / T_);

    const float cur  = hidden[i];
    const float prev = (t == 0) ? sx[(size_t)b * H_ + h] : hidden[i - H_];
    const float mk = tmk[h], mv = tmv[h], mr = tmr[h];
    mixK[i] = (__bf16)(cur * mk + prev * (1.0f - mk));
    mixV[i] = (__bf16)(cur * mv + prev * (1.0f - mv));
    mixR[i] = (__bf16)(cur * mr + prev * (1.0f - mr));
}

// ---------------------------------------------------------------------------
// C[M,N](f32) = A[M,K](bf16, row-major) x Bt[N,K](bf16, row-major)^T
// Wave tile 32x64 (2x4 WMMA accumulators), block = 8 waves -> 64x256 tile.
// Fragment loads follow the CDNA5 ISA VGPR layouts (ISA 7.12.2).
// ---------------------------------------------------------------------------
__global__ void __launch_bounds__(256)
gemm_bf16_kernel(const __bf16* __restrict__ A, const __bf16* __restrict__ Bt,
                 float* __restrict__ C, int M, int N, int K)
{
    const int lane = threadIdx.x & 31;
    const int wave = threadIdx.x >> 5;           // 0..7
    const int wm   = wave & 1;                   // M direction
    const int wn   = wave >> 1;                  // N direction
    const int waveM = blockIdx.x * 64  + wm * 32;
    const int waveN = blockIdx.y * 256 + wn * 64;

    const int lr  = lane & 15;                   // row (A) / col (B) within tile
    const int kha = (lane >> 4) * 8;             // A K sub-offset: 0 or 8
    const int khb = (lane >> 4) * 8;             // B K sub-offset: 0 or 8

    const __bf16* aRow0 = A + (size_t)(waveM + lr)      * K;
    const __bf16* aRow1 = A + (size_t)(waveM + 16 + lr) * K;
    const __bf16* bRow[4];
#pragma unroll
    for (int n = 0; n < 4; ++n)
        bRow[n] = Bt + (size_t)(waveN + n * 16 + lr) * K;

    v8f acc[2][4] = {};

    for (int ks = 0; ks < K; ks += 32) {
        const int ka = ks + kha;
        const int kb = ks + khb;

        __builtin_prefetch(aRow0 + ka + 32, 0, 3);
        __builtin_prefetch(bRow[0] + kb + 32, 0, 3);

        Frag16 a0, a1, b[4];
        a0.u[0] = *(const v4u*)(aRow0 + ka);
        a0.u[1] = *(const v4u*)(aRow0 + ka + 16);
        a1.u[0] = *(const v4u*)(aRow1 + ka);
        a1.u[1] = *(const v4u*)(aRow1 + ka + 16);
#pragma unroll
        for (int n = 0; n < 4; ++n) {
            b[n].u[0] = *(const v4u*)(bRow[n] + kb);
            b[n].u[1] = *(const v4u*)(bRow[n] + kb + 16);
        }
#pragma unroll
        for (int n = 0; n < 4; ++n) {
            acc[0][n] = __builtin_amdgcn_wmma_f32_16x16x32_bf16(
                false, a0.v, false, b[n].v, (short)0, acc[0][n], false, false);
            acc[1][n] = __builtin_amdgcn_wmma_f32_16x16x32_bf16(
                false, a1.v, false, b[n].v, (short)0, acc[1][n], false, false);
        }
    }

    // C/D layout: VGPR r holds M = r + 8*(lane/16), N = lane%16
    const int rowOfs = (lane >> 4) * 8;
    const int col    = lr;
#pragma unroll
    for (int m = 0; m < 2; ++m)
#pragma unroll
        for (int n = 0; n < 4; ++n) {
            float* cp = C + (size_t)(waveM + m * 16 + rowOfs) * N
                          + (waveN + n * 16 + col);
#pragma unroll
            for (int r = 0; r < 8; ++r)
                cp[(size_t)r * N] = acc[m][n][r];
        }
}

// ---------------------------------------------------------------------------
// WKV chunked scan. Unnormalized state (a^, b^) with a^ = aa*e^pp obeys
//   a^' = e^decay * a^ + e^k * v      (linear per step)
// so a length-L segment is the affine map  a^_out = e^{L*decay}*a^_in + Sum,
// where Sum is the reference update run from the empty state (0,0,-1e30).
//
// Pass A: per (s,b,h) segment summary (sa,sb,sp), max-normalized.
// ---------------------------------------------------------------------------
__global__ void __launch_bounds__(256)
wkv_seg_summary_kernel(const float* __restrict__ kbuf,
                       const float* __restrict__ vbuf,
                       const float* __restrict__ time_decay,
                       float* __restrict__ segA, float* __restrict__ segB,
                       float* __restrict__ segP)
{
    const int tid = blockIdx.x * blockDim.x + threadIdx.x;   // over S*B*H
    const int h   = tid % H_;
    const int tmp = tid / H_;
    const int b   = tmp % B_;
    const int s   = tmp / B_;

    const float decay = -__expf(time_decay[h]);

    float sa = 0.0f, sb = 0.0f, sp = -1e30f;
    const size_t base = ((size_t)b * T_ + (size_t)s * L_) * H_ + h;
    for (int i = 0; i < L_; ++i) {
        const size_t off = base + (size_t)i * H_;
        const float kk = kbuf[off];
        const float vv = vbuf[off];
        const float ww = decay + sp;
        const float q  = fmaxf(ww, kk);
        const float e1 = __expf(ww - q);
        const float e2 = __expf(kk - q);
        sa = e1 * sa + e2 * vv;
        sb = e1 * sb + e2;
        sp = q;
    }
    segA[tid] = sa;
    segB[tid] = sb;
    segP[tid] = sp;
}

// ---------------------------------------------------------------------------
// Pass B: sequential combine over the S segments (per (b,h) channel):
//   state <- e^{L*decay} * state + segment   (max-normalized)
// Records the prefix state at each segment start; emits final aa/bb/pp and
// the last hidden token.
// ---------------------------------------------------------------------------
__global__ void __launch_bounds__(256)
wkv_seg_prefix_kernel(const float* __restrict__ segA,
                      const float* __restrict__ segB,
                      const float* __restrict__ segP,
                      const float* __restrict__ aa_in,
                      const float* __restrict__ bb_in,
                      const float* __restrict__ pp_in,
                      const float* __restrict__ time_decay,
                      const float* __restrict__ hidden,
                      float* __restrict__ preA, float* __restrict__ preB,
                      float* __restrict__ preP,
                      float* __restrict__ out_aa, float* __restrict__ out_bb,
                      float* __restrict__ out_pp, float* __restrict__ out_last)
{
    const int tid = blockIdx.x * blockDim.x + threadIdx.x;   // over B*H
    const int b = tid / H_;
    const int h = tid % H_;
    const size_t BH = (size_t)B_ * H_;

    const float Ldecay = (float)L_ * (-__expf(time_decay[h]));

    float aa = aa_in[tid];
    float bb = bb_in[tid];
    float pp = pp_in[tid];

    for (int s = 0; s < S_; ++s) {
        const size_t si = (size_t)s * BH + tid;
        preA[si] = aa; preB[si] = bb; preP[si] = pp;   // state at segment start

        const float sa = segA[si], sb = segB[si], sp = segP[si];
        const float wwp = Ldecay + pp;
        const float q   = fmaxf(wwp, sp);
        const float e1  = __expf(wwp - q);
        const float e2  = __expf(sp  - q);
        aa = e1 * aa + e2 * sa;
        bb = e1 * bb + e2 * sb;
        pp = q;
    }
    out_aa[tid] = aa;
    out_bb[tid] = bb;
    out_pp[tid] = pp;
    out_last[tid] = hidden[((size_t)b * T_ + (T_ - 1)) * H_ + h];
}

// ---------------------------------------------------------------------------
// Pass C: replay each segment from its prefix state; per-token math identical
// to the reference. Fuses sigmoid(r)*c -> bf16 A-matrix for the output GEMM.
// ---------------------------------------------------------------------------
__global__ void __launch_bounds__(256)
wkv_seg_emit_kernel(const float* __restrict__ kbuf,
                    const float* __restrict__ vbuf,
                    const float* __restrict__ rraw,
                    const float* __restrict__ preA,
                    const float* __restrict__ preB,
                    const float* __restrict__ preP,
                    const float* __restrict__ time_decay,
                    const float* __restrict__ time_first,
                    __bf16* __restrict__ rc)
{
    const int tid = blockIdx.x * blockDim.x + threadIdx.x;   // over S*B*H
    const int h   = tid % H_;
    const int tmp = tid / H_;
    const int b   = tmp % B_;
    const int s   = tmp / B_;

    const float tf    = time_first[h];
    const float decay = -__expf(time_decay[h]);

    float aa = preA[tid], bb = preB[tid], pp = preP[tid];

    const size_t base = ((size_t)b * T_ + (size_t)s * L_) * H_ + h;
    for (int i = 0; i < L_; ++i) {
        const size_t off = base + (size_t)i * H_;
        const float kk = kbuf[off];
        const float vv = vbuf[off];
        const float rr = rraw[off];

        const float ww = tf + kk;
        const float q  = fmaxf(pp, ww);
        const float e1 = __expf(pp - q);
        const float e2 = __expf(ww - q);
        const float c  = (e1 * aa + e2 * vv) / (e1 * bb + e2);

        const float ww2 = decay + pp;
        const float q2  = fmaxf(ww2, kk);
        const float f1  = __expf(ww2 - q2);
        const float f2  = __expf(kk - q2);
        aa = f1 * aa + f2 * vv;
        bb = f1 * bb + f2;
        pp = q2;

        const float r = 1.0f / (1.0f + __expf(-rr));
        rc[off] = (__bf16)(r * c);
    }
}

// ---------------------------------------------------------------------------
extern "C" void kernel_launch(void* const* d_in, const int* in_sizes, int n_in,
                              void* d_out, int out_size, void* d_ws, size_t ws_size,
                              hipStream_t stream)
{
    (void)in_sizes; (void)n_in; (void)out_size; (void)ws_size;

    const float* hidden = (const float*)d_in[0];
    const float* sx     = (const float*)d_in[1];
    const float* aa     = (const float*)d_in[2];
    const float* bb     = (const float*)d_in[3];
    const float* pp     = (const float*)d_in[4];
    const float* tdec   = (const float*)d_in[5];
    const float* tfir   = (const float*)d_in[6];
    const float* tmk    = (const float*)d_in[7];
    const float* tmv    = (const float*)d_in[8];
    const float* tmr    = (const float*)d_in[9];
    const float* Wk     = (const float*)d_in[10];
    const float* Wv     = (const float*)d_in[11];
    const float* Wr     = (const float*)d_in[12];
    const float* Wo     = (const float*)d_in[13];

    const size_t BTH = (size_t)B_ * T_ * H_;     // 16,777,216
    const size_t BH  = (size_t)B_ * H_;          // 8,192
    const size_t HH  = (size_t)H_ * H_;          // 4,194,304
    const size_t SBH = (size_t)S_ * BH;          // 131,072

    // workspace carve-up (~372 MB)
    char* ws = (char*)d_ws;
    size_t o = 0;
    __bf16* mixK = (__bf16*)(ws + o); o += BTH * 2;
    __bf16* mixV = (__bf16*)(ws + o); o += BTH * 2;
    __bf16* mixR = (__bf16*)(ws + o); o += BTH * 2;
    __bf16* rcA  = (__bf16*)(ws + o); o += BTH * 2;
    __bf16* WtK  = (__bf16*)(ws + o); o += HH * 2;
    __bf16* WtV  = (__bf16*)(ws + o); o += HH * 2;
    __bf16* WtR  = (__bf16*)(ws + o); o += HH * 2;
    __bf16* WtO  = (__bf16*)(ws + o); o += HH * 2;
    float*  kbuf = (float*)(ws + o);  o += BTH * 4;
    float*  vbuf = (float*)(ws + o);  o += BTH * 4;
    float*  rraw = (float*)(ws + o);  o += BTH * 4;
    float*  segA = (float*)(ws + o);  o += SBH * 4;
    float*  segB = (float*)(ws + o);  o += SBH * 4;
    float*  segP = (float*)(ws + o);  o += SBH * 4;
    float*  preA = (float*)(ws + o);  o += SBH * 4;
    float*  preB = (float*)(ws + o);  o += SBH * 4;
    float*  preP = (float*)(ws + o);  o += SBH * 4;

    // output carve-up: out[B,T,H], hidden_last[B,H], aa, bb, pp
    float* out      = (float*)d_out;
    float* out_last = out + BTH;
    float* out_aa   = out_last + BH;
    float* out_bb   = out_aa + BH;
    float* out_pp   = out_bb + BH;

    // 1. transpose + bf16-convert the four weights
    wt_transpose_kernel<<<dim3(H_ / 32, H_ / 32, 4), dim3(32, 8), 0, stream>>>(
        Wk, Wv, Wr, Wo, WtK, WtV, WtR, WtO);

    // 2. token-shift mix -> bf16 GEMM inputs
    mix_kernel<<<(unsigned)(BTH / 256), 256, 0, stream>>>(
        hidden, sx, tmk, tmv, tmr, mixK, mixV, mixR);

    // 3. k / v / r projections (WMMA bf16, fp32 accumulate)
    const dim3 ggrid(M_ / 64, H_ / 256);
    gemm_bf16_kernel<<<ggrid, 256, 0, stream>>>(mixK, WtK, kbuf, M_, H_, H_);
    gemm_bf16_kernel<<<ggrid, 256, 0, stream>>>(mixV, WtV, vbuf, M_, H_, H_);
    gemm_bf16_kernel<<<ggrid, 256, 0, stream>>>(mixR, WtR, rraw, M_, H_, H_);

    // 4. chunked WKV scan: summaries -> prefix combine -> replay
    wkv_seg_summary_kernel<<<(unsigned)(SBH / 256), 256, 0, stream>>>(
        kbuf, vbuf, tdec, segA, segB, segP);
    wkv_seg_prefix_kernel<<<(unsigned)(BH / 256), 256, 0, stream>>>(
        segA, segB, segP, aa, bb, pp, tdec, hidden,
        preA, preB, preP, out_aa, out_bb, out_pp, out_last);
    wkv_seg_emit_kernel<<<(unsigned)(SBH / 256), 256, 0, stream>>>(
        kbuf, vbuf, rraw, preA, preB, preP, tdec, tfir, rcA);

    // 5. output projection
    gemm_bf16_kernel<<<ggrid, 256, 0, stream>>>(rcA, WtO, out, M_, H_, H_);
}
